// LinearAttention_2637109919827
// MI455X (gfx1250) — matrix operand order repaired and checked by reading, not compiled, not run on hardware
//
#include <hip/hip_runtime.h>

typedef __attribute__((ext_vector_type(16))) _Float16 v16h;
typedef __attribute__((ext_vector_type(4)))  _Float16 v4h;
typedef __attribute__((ext_vector_type(8)))  float    v8f;

#define HEADS    4
#define DIM_HEAD 32
#define HIDDEN   128
#define CIN      256
#define QKV_ROWS 384
#define NPIX     4096
#define BATCH    32
#define SCALE_Q  0.17677669529663687f  /* 32^-0.5 */
#define EPS_GN   1e-5f

// ---- WMMA fragment index helpers (contiguous k order, same for A and B) ----
// A-matrix 16x32 f16: lane holds row M=lane&15, half=lane>>4,
// element e -> K = half*16 + e   (contiguous, matches B)
__device__ __forceinline__ int kA(int e, int half) {
  return half * 16 + e;
}
// B-matrix 32x16 f16: lane holds col N=lane&15, element e -> K = half*16 + e
// With B tiles stored transposed in LDS ([n][k]), a lane's 16 values are
// contiguous -> 2x ds_load_b128 per fragment.

// ---------------------------------------------------------------------------
__global__ void la_zero_kernel(float* __restrict__ p, int n) {
  int i = blockIdx.x * blockDim.x + threadIdx.x;
  if (i < n) p[i] = 0.0f;
}

// ---------------------------------------------------------------------------
// Kernel 1: per (batch, 16-pixel tile):
//   qkv[384,16] = Wqkv[384,256] x x[b][256,16] + bqkv   (WMMA f16, f32 acc)
//   softmax over dim_head for q (store *SCALE to ws f16) and k (in LDS)
//   ctx[b,h,:,:] += k_tile . v_tile^T   (atomic f32)
__global__ __launch_bounds__(256) void la_qkv_kernel(
    const float* __restrict__ x, const float* __restrict__ Wqkv,
    const float* __restrict__ bqkv, _Float16* __restrict__ q_soft,
    float* __restrict__ ctx)
{
  __shared__ __align__(16) _Float16 sW[QKV_ROWS * 32];   // 24 KB  A chunk [o][k]
  __shared__ __align__(16) _Float16 sX[16 * 32];         //  1 KB  B transposed [n][k]
  __shared__ __align__(16) float    sQ[QKV_ROWS * 16];   // 24 KB  qkv result tile

  const int tid  = threadIdx.x;
  const int b    = blockIdx.x >> 8;
  const int n0   = (blockIdx.x & 255) * 16;
  const int lane = tid & 31;
  const int wave = tid >> 5;
  const int row  = lane & 15;
  const int half = lane >> 4;

  v8f acc[3];
  v8f zf = {};
#pragma unroll
  for (int t = 0; t < 3; ++t) acc[t] = zf;

  for (int k0 = 0; k0 < CIN; k0 += 32) {
    // stage A: 12288 halfs as float4 loads (12 per thread)
#pragma unroll
    for (int j = 0; j < 12; ++j) {
      int idx = tid + j * 256;            // 3072 float4 groups
      int o = idx >> 3, kk = (idx & 7) * 4;
      float4 w = *(const float4*)&Wqkv[o * CIN + k0 + kk];
      v4h p; p[0] = (_Float16)w.x; p[1] = (_Float16)w.y;
      p[2] = (_Float16)w.z; p[3] = (_Float16)w.w;
      *(v4h*)&sW[o * 32 + kk] = p;
    }
    // stage B (transposed): 512 halfs, float4 loads along n, scatter to [n][k]
    for (int i = tid; i < 128; i += 256) {
      int kk = i >> 2, nn = (i & 3) * 4;
      float4 xv = *(const float4*)&x[((size_t)b * CIN + (size_t)(k0 + kk)) * NPIX + n0 + nn];
      sX[(nn + 0) * 32 + kk] = (_Float16)xv.x;
      sX[(nn + 1) * 32 + kk] = (_Float16)xv.y;
      sX[(nn + 2) * 32 + kk] = (_Float16)xv.z;
      sX[(nn + 3) * 32 + kk] = (_Float16)xv.w;
    }
    __syncthreads();

    v16h bf;
#pragma unroll
    for (int e = 0; e < 16; ++e) bf[e] = sX[row * 32 + half * 16 + e];

#pragma unroll
    for (int t = 0; t < 3; ++t) {
      const int m0 = (wave * 3 + t) * 16;   // 8 waves x 3 tiles = 24 M-tiles
      v16h af;
#pragma unroll
      for (int e = 0; e < 16; ++e) af[e] = sW[(m0 + row) * 32 + kA(e, half)];
      acc[t] = __builtin_amdgcn_wmma_f32_16x16x32_f16(
          false, af, false, bf, (short)0, acc[t], false, false);
    }
    __syncthreads();
  }

  // write qkv tile (+bias) to LDS
#pragma unroll
  for (int t = 0; t < 3; ++t) {
    const int m0 = (wave * 3 + t) * 16;
#pragma unroll
    for (int v = 0; v < 8; ++v) {
      int r = m0 + v + half * 8;
      sQ[r * 16 + row] = acc[t][v] + bqkv[r];
    }
  }
  __syncthreads();

  // softmax over dim_head for q (threads 0..63) and k (threads 64..127)
  if (tid < 128) {
    const int isK = tid >> 6;
    const int h   = (tid >> 4) & 3;
    const int nn  = tid & 15;
    const int rb  = isK * 128 + h * 32;
    float m = -1e30f;
#pragma unroll
    for (int d = 0; d < 32; ++d) m = fmaxf(m, sQ[(rb + d) * 16 + nn]);
    float ex[32];
    float s = 0.0f;
#pragma unroll
    for (int d = 0; d < 32; ++d) { ex[d] = expf(sQ[(rb + d) * 16 + nn] - m); s += ex[d]; }
    float inv = 1.0f / s;
    if (isK == 0) {
#pragma unroll
      for (int d = 0; d < 32; ++d) {
        float qv = ex[d] * inv * SCALE_Q;
        q_soft[((size_t)(b * HEADS + h) * DIM_HEAD + d) * NPIX + n0 + nn] = (_Float16)qv;
      }
    } else {
#pragma unroll
      for (int d = 0; d < 32; ++d) sQ[(rb + d) * 16 + nn] = ex[d] * inv;
    }
  }
  __syncthreads();

  // partial context: ctx[h][d][e] += sum_nn k[h,d,nn] * v[h,e,nn]
  for (int t = tid; t < HEADS * DIM_HEAD * DIM_HEAD; t += 256) {
    int h = t >> 10, d = (t >> 5) & 31, e = t & 31;
    float s = 0.0f;
#pragma unroll
    for (int nn = 0; nn < 16; ++nn)
      s += sQ[(128 + h * 32 + d) * 16 + nn] * sQ[(256 + h * 32 + e) * 16 + nn];
    atomicAdd(&ctx[((size_t)(b * HEADS + h) * DIM_HEAD + d) * DIM_HEAD + e], s);
  }
}

// ---------------------------------------------------------------------------
// Kernel 2: per batch fold context into output projection:
//   M[co, h*32+d] = sum_e Wout[co, h*32+e] * ctx[b,h,d,e]     (WMMA)
__global__ __launch_bounds__(256) void la_combine_kernel(
    const float* __restrict__ Wout, const float* __restrict__ ctx,
    _Float16* __restrict__ Mf16)
{
  __shared__ __align__(16) _Float16 sA[256 * 32];  // 16 KB  [co][e]
  __shared__ __align__(16) _Float16 sB[32 * 32];   //  2 KB  transposed [n=d][k=e]
  const int tid  = threadIdx.x;
  const int b    = blockIdx.x;
  const int lane = tid & 31;
  const int wave = tid >> 5;
  const int row  = lane & 15;
  const int half = lane >> 4;

  for (int h = 0; h < HEADS; ++h) {
    // stage A: 8192 halfs as 2048 float4 loads (8 per thread)
#pragma unroll
    for (int j = 0; j < 8; ++j) {
      int idx = tid + j * 256;
      int o = idx >> 3, e4 = (idx & 7) * 4;
      float4 w = *(const float4*)&Wout[o * HIDDEN + h * 32 + e4];
      v4h p; p[0] = (_Float16)w.x; p[1] = (_Float16)w.y;
      p[2] = (_Float16)w.z; p[3] = (_Float16)w.w;
      *(v4h*)&sA[o * 32 + e4] = p;
    }
    // stage B: ctx[h][d][e] is already [n=d][k=e] layout -> direct copy
    for (int i = tid; i < 256; i += 256) {
      int d = i >> 3, e4 = (i & 7) * 4;
      float4 c4 = *(const float4*)&ctx[((size_t)(b * HEADS + h) * DIM_HEAD + d) * DIM_HEAD + e4];
      v4h p; p[0] = (_Float16)c4.x; p[1] = (_Float16)c4.y;
      p[2] = (_Float16)c4.z; p[3] = (_Float16)c4.w;
      *(v4h*)&sB[d * 32 + e4] = p;
    }
    __syncthreads();
#pragma unroll
    for (int tt = 0; tt < 4; ++tt) {
      int tile = wave + tt * 8;          // 32 tiles: 16 co-tiles x 2 d-tiles
      int ct = tile >> 1, dt = tile & 1;
      v16h af, bf;
#pragma unroll
      for (int e = 0; e < 16; ++e) af[e] = sA[(ct * 16 + row) * 32 + kA(e, half)];
#pragma unroll
      for (int e = 0; e < 16; ++e) bf[e] = sB[(dt * 16 + row) * 32 + half * 16 + e];
      v8f c = {};
      c = __builtin_amdgcn_wmma_f32_16x16x32_f16(
          false, af, false, bf, (short)0, c, false, false);
#pragma unroll
      for (int v = 0; v < 8; ++v) {
        int co = ct * 16 + v + half * 8;
        int dd = dt * 16 + row;
        Mf16[((size_t)b * 256 + co) * HIDDEN + h * 32 + dd] = (_Float16)c[v];
      }
    }
    __syncthreads();
  }
}

// ---------------------------------------------------------------------------
// Kernel 3: out[b,256,4096] = M_b[256,128] x q_soft[b][128,4096] + bout (WMMA)
// plus per-batch sum / sumsq accumulation for GroupNorm.
// A-tile staging uses CDNA5 async global->LDS copies (ASYNCcnt).
__global__ __launch_bounds__(256) void la_proj_kernel(
    const _Float16* __restrict__ Mf16, const _Float16* __restrict__ q_soft,
    const float* __restrict__ bout, float* __restrict__ out,
    float* __restrict__ stats)
{
  __shared__ __align__(16) _Float16 sA[128 * 32];  // 8 KB  [co][k]
  __shared__ __align__(16) _Float16 sB[64 * 32];   // 4 KB  transposed [n][k]
  __shared__ __align__(16) float    red[512];      // 2 KB
  const int tid  = threadIdx.x;
  const int b    = blockIdx.x >> 7;
  const int rem  = blockIdx.x & 127;
  const int co0  = (rem >> 6) * 128;
  const int n0   = (rem & 63) * 64;
  const int lane = tid & 31;
  const int wave = tid >> 5;
  const int row  = lane & 15;
  const int half = lane >> 4;

  v8f acc[4];
  v8f zf = {};
#pragma unroll
  for (int i = 0; i < 4; ++i) acc[i] = zf;

  for (int k0 = 0; k0 < HIDDEN; k0 += 32) {
    // stage A: 4096 halfs as async global->LDS B64 copies (4 per thread).
    // Each lane: 8 aligned bytes, LDS dest = low 32 bits of generic pointer.
#pragma unroll
    for (int j = 0; j < 4; ++j) {
      int idx = tid + j * 256;            // 1024 groups of 4 halfs
      int r = idx >> 3, kk = (idx & 7) * 4;
      const _Float16* gsrc =
          &Mf16[((size_t)b * 256 + co0 + r) * HIDDEN + k0 + kk];
      unsigned ldst = (unsigned)(uintptr_t)(void*)&sA[r * 32 + kk];
      asm volatile("global_load_async_to_lds_b64 %0, %1, off"
                   :: "v"(ldst), "v"(gsrc) : "memory");
    }
    // stage B (transposed): 2048 halfs, 8-byte loads along n, scatter [n][k]
#pragma unroll
    for (int j = 0; j < 2; ++j) {
      int idx = tid + j * 256;            // 512 groups of 4 halfs
      int kk = idx >> 4, nn = (idx & 15) * 4;
      v4h qv = *(const v4h*)&q_soft[((size_t)b * HIDDEN + (size_t)(k0 + kk)) * NPIX + n0 + nn];
      sB[(nn + 0) * 32 + kk] = qv[0];
      sB[(nn + 1) * 32 + kk] = qv[1];
      sB[(nn + 2) * 32 + kk] = qv[2];
      sB[(nn + 3) * 32 + kk] = qv[3];
    }
    // wait for this wave's async copies, then workgroup barrier
    asm volatile("s_wait_asynccnt 0x0" ::: "memory");
    __syncthreads();
    v16h af;
#pragma unroll
    for (int e = 0; e < 16; ++e) af[e] = sA[(wave * 16 + row) * 32 + kA(e, half)];
#pragma unroll
    for (int ns = 0; ns < 4; ++ns) {
      v16h bf;
#pragma unroll
      for (int e = 0; e < 16; ++e) bf[e] = sB[(ns * 16 + row) * 32 + half * 16 + e];
      acc[ns] = __builtin_amdgcn_wmma_f32_16x16x32_f16(
          false, af, false, bf, (short)0, acc[ns], false, false);
    }
    __syncthreads();
  }

  float lsum = 0.0f, lsq = 0.0f;
#pragma unroll
  for (int ns = 0; ns < 4; ++ns) {
#pragma unroll
    for (int v = 0; v < 8; ++v) {
      int co = co0 + wave * 16 + v + half * 8;
      int n  = n0 + ns * 16 + row;
      float val = acc[ns][v] + bout[co];
      out[((size_t)b * 256 + co) * NPIX + n] = val;
      lsum += val;
      lsq  += val * val;
    }
  }
  red[tid] = lsum;
  red[256 + tid] = lsq;
  __syncthreads();
  for (int s = 128; s > 0; s >>= 1) {
    if (tid < s) {
      red[tid]       += red[tid + s];
      red[256 + tid] += red[256 + tid + s];
    }
    __syncthreads();
  }
  if (tid == 0) {
    atomicAdd(&stats[b * 2 + 0], red[0]);
    atomicAdd(&stats[b * 2 + 1], red[256]);
  }
}

// ---------------------------------------------------------------------------
// Kernel 4: GroupNorm(1, C) per sample + affine
__global__ void la_gnorm_kernel(float* __restrict__ out,
                                const float* __restrict__ stats,
                                const float* __restrict__ gamma,
                                const float* __restrict__ beta)
{
  size_t i = (size_t)blockIdx.x * blockDim.x + threadIdx.x;
  const size_t per_b = (size_t)256 * NPIX;
  const size_t total = (size_t)BATCH * per_b;
  if (i >= total) return;
  int b = (int)(i / per_b);
  int c = (int)((i / NPIX) & 255);
  float n    = (float)per_b;
  float mean = stats[b * 2 + 0] / n;
  float var  = stats[b * 2 + 1] / n - mean * mean;
  float inv  = rsqrtf(var + EPS_GN);
  float v    = out[i];
  out[i] = (v - mean) * inv * gamma[c] + beta[c];
}

// ---------------------------------------------------------------------------
extern "C" void kernel_launch(void* const* d_in, const int* in_sizes, int n_in,
                              void* d_out, int out_size, void* d_ws, size_t ws_size,
                              hipStream_t stream) {
  const float* x     = (const float*)d_in[0];
  const float* Wqkv  = (const float*)d_in[1];
  const float* bqkv  = (const float*)d_in[2];
  const float* Wout  = (const float*)d_in[3];
  const float* bout  = (const float*)d_in[4];
  const float* gamma = (const float*)d_in[5];
  const float* beta  = (const float*)d_in[6];
  float* out = (float*)d_out;

  // workspace layout (~36.2 MB)
  char* ws = (char*)d_ws;
  _Float16* q_soft = (_Float16*)(ws);                    // 32*128*4096 f16 = 33554432 B
  float*    ctx    = (float*)   (ws + 33554432);         // 32*4*32*32  f32 =   524288 B
  _Float16* Mf16   = (_Float16*)(ws + 34078720);         // 32*256*128  f16 =  2097152 B
  float*    stats  = (float*)   (ws + 36175872);         // 64 f32

  // zero accumulators every call (harness does not re-poison between replays)
  la_zero_kernel<<<(BATCH * HEADS * DIM_HEAD * DIM_HEAD + 255) / 256, 256, 0, stream>>>(
      ctx, BATCH * HEADS * DIM_HEAD * DIM_HEAD);
  la_zero_kernel<<<1, 64, 0, stream>>>(stats, BATCH * 2);

  // 1) qkv GEMM + softmax + context accumulation
  la_qkv_kernel<<<BATCH * (NPIX / 16), 256, 0, stream>>>(x, Wqkv, bqkv, q_soft, ctx);

  // 2) fold context into Wout: per-batch M[256,128]
  la_combine_kernel<<<BATCH, 256, 0, stream>>>(Wout, ctx, Mf16);

  // 3) final GEMM + bias + GroupNorm stats
  la_proj_kernel<<<BATCH * 2 * (NPIX / 64), 256, 0, stream>>>(Mf16, q_soft, bout, out, stats);

  // 4) GroupNorm apply
  la_gnorm_kernel<<<(unsigned)(((size_t)BATCH * 256 * NPIX + 255) / 256), 256, 0, stream>>>(
      out, stats, gamma, beta);
}
